// DAriEL_Encoder_Cell_2_86586540687905
// MI455X (gfx1250) — matrix operand order repaired and checked by reading, not compiled, not run on hardware
//
#include <hip/hip_runtime.h>

// ---------------------------------------------------------------------------
// DAriEL_Encoder_Cell_2 for MI455X (gfx1250, wave32).
//
//   * z[:,0,:] == 0 exactly (rng == 0 at t==0).
//   * z[b,t,:] = 3*cl + 1.5*softmax(h_t)[s], s = input_tokens[b,t] (scalar
//     broadcast over LAT).
//   * Nested LM prefixes -> ONE incremental LSTM pass (7 effective steps).
//
// Per step: gates[32,8192] = x@Wi + h@Wh + b via v_wmma_f32_16x16x32_f16
// (f16 in, f32 accumulate). Wh phase uses double-buffered
// global_load_async_to_lds_b128 staging (ASYNCcnt) so the DMA of K-slab s+1
// overlaps the WMMA of slab s; weights are L2-resident across the 7 steps.
// ---------------------------------------------------------------------------

typedef __attribute__((ext_vector_type(16))) _Float16 v16h;
typedef __attribute__((ext_vector_type(8)))  float    v8f;

#define BATCHN 32
#define EMBD   256
#define HID    2048
#define NGATE  (4 * HID)
#define SEQT   8
#define LATN   128
#define UT     64            // hidden units per block -> 256 gate columns

#define BROW   260           // f32 row stride of the B slab in LDS (16B-aligned rows)
#define SLABB  (32 * BROW)   // one B slab: 32 K-rows x 256 cols (+pad)
#define AROW   36            // f32 row stride of the A slab in LDS
#define SLABA  (32 * AROW)   // one A slab: 32 batch rows x 32 K

// ---- CDNA5 async copy: global -> LDS, 16B per lane, tracked by ASYNCcnt ---
__device__ __forceinline__ void async_copy16(unsigned int lds_off, const float* g) {
  asm volatile("global_load_async_to_lds_b128 %0, %1, off"
               :: "v"(lds_off), "v"((unsigned long long)(uintptr_t)g)
               : "memory");
}

#if defined(__has_builtin) && __has_builtin(__builtin_amdgcn_s_wait_asynccnt)
#define WAIT_ASYNC(n) __builtin_amdgcn_s_wait_asynccnt(n)
#else
#define WAIT_ASYNC(n) asm volatile("s_wait_asynccnt %0" :: "i"(n) : "memory")
#endif

// ---------------------------------------------------------------------------
// Init: zero h0, c, and the t==0 output slice.
// ---------------------------------------------------------------------------
__global__ void dariel_init_kernel(float* __restrict__ h0,
                                   float* __restrict__ c,
                                   float* __restrict__ out) {
  int i = blockIdx.x * blockDim.x + threadIdx.x;
  if (i < BATCHN * HID) {
    h0[i] = 0.0f;
    c[i]  = 0.0f;
  }
  if (i < BATCHN * LATN) {
    int b = i / LATN, l = i % LATN;
    out[(size_t)b * SEQT * LATN + l] = 0.0f;   // t = 0 slice
  }
}

// ---------------------------------------------------------------------------
// One LSTM step. Block = 256 threads (8 waves), handles UT=64 hidden units
// (256 gate columns). Wave w owns output cols [32w, 32w+32), both 16-row
// batch tiles => 4 WMMA accumulators per wave.
// ---------------------------------------------------------------------------
__global__ void dariel_lstm_step_kernel(const int*   __restrict__ tok,
                                        const float* __restrict__ E,
                                        const float* __restrict__ Wi,
                                        const float* __restrict__ Wh,
                                        const float* __restrict__ bias,
                                        const float* __restrict__ hin,
                                        float*       __restrict__ hout,
                                        float*       __restrict__ c,
                                        int t) {
  __shared__ float smemB[2 * SLABB];          // 65 KB, double-buffered Wh slab
  __shared__ float smemA[2 * SLABA];          // 9 KB, double-buffered h slab
  float* gsm = smemB;                         // 32x256 gate spill aliases smemB

  const int w    = threadIdx.x >> 5;
  const int lane = threadIdx.x & 31;
  const int l15  = lane & 15;
  const int hi16 = (lane & 16) ? 1 : 0;
  const int u0   = blockIdx.x * UT;

  // A-layout (16-bit 16x32): M = lane&15; elem e -> K = kbaseA + e + (e & 8).
  const int kbaseA = hi16 ? 8 : 0;
  // B-layout (16-bit 32x16): N = lane&15; elem e -> K = kbaseB + e + (e & 8).
  const int kbaseB = hi16 ? 8 : 0;

  // Local gate column lc in [0,256): gate = lc>>6, unit = u0 + (lc&63).
  int lc[2], wcol[2];
#pragma unroll
  for (int ct = 0; ct < 2; ++ct) {
    lc[ct]   = w * 32 + ct * 16 + l15;
    wcol[ct] = (lc[ct] >> 6) * HID + u0 + (lc[ct] & 63);
  }

  int tokm[2];
#pragma unroll
  for (int rt = 0; rt < 2; ++rt)
    tokm[rt] = tok[(rt * 16 + l15) * SEQT + (t - 1)];

  v8f acc[2][2] = {};

  // ---- Phase 1: x @ Wi, x = E[token], K = EMBD (direct, 8 iters) ----------
  for (int k0 = 0; k0 < EMBD; k0 += 32) {
    v16h a[2];
#pragma unroll
    for (int rt = 0; rt < 2; ++rt) {
      const float* xrow = E + (size_t)tokm[rt] * EMBD;
#pragma unroll
      for (int e = 0; e < 16; ++e)
        a[rt][e] = (_Float16)xrow[k0 + kbaseA + e + (e & 8)];
    }
    v16h bm[2];
#pragma unroll
    for (int ct = 0; ct < 2; ++ct)
#pragma unroll
      for (int e = 0; e < 16; ++e)
        bm[ct][e] = (_Float16)Wi[(size_t)(k0 + kbaseB + e + (e & 8)) * NGATE + wcol[ct]];
#pragma unroll
    for (int rt = 0; rt < 2; ++rt)
#pragma unroll
      for (int ct = 0; ct < 2; ++ct)
        acc[rt][ct] = __builtin_amdgcn_wmma_f32_16x16x32_f16(
            false, a[rt], false, bm[ct], (short)0, acc[rt][ct], false, false);
  }

  // ---- Phase 2: h @ Wh, K = HID, async double-buffered LDS staging --------
  // Stage one 32-K slab: B = 32x256 f32 (2048 x 16B), A = 32x32 f32 (256 x 16B).
  // 9 async b128 per thread per slab.
  auto stageSlab = [&](int k0, int buf) {
#pragma unroll
    for (int i = 0; i < 8; ++i) {
      int idx  = threadIdx.x + i * 256;       // 0..2047
      int k    = idx >> 6;                    // K-row within slab
      int f    = idx & 63;                    // float4 within row
      int gate = f >> 4;
      int c4   = (f & 15) << 2;
      const float* g = Wh + (size_t)(k0 + k) * NGATE + gate * HID + u0 + c4;
      unsigned int l = (unsigned int)(uintptr_t)
          &smemB[buf * SLABB + k * BROW + gate * 64 + c4];
      async_copy16(l, g);
    }
    {
      int m = threadIdx.x >> 3;
      int f = threadIdx.x & 7;
      const float* g = hin + (size_t)m * HID + k0 + f * 4;
      unsigned int l = (unsigned int)(uintptr_t)
          &smemA[buf * SLABA + m * AROW + f * 4];
      async_copy16(l, g);
    }
  };

  stageSlab(0, 0);
  for (int s = 0; s < HID / 32; ++s) {
    const int buf = s & 1;
    if (s + 1 < HID / 32) {
      stageSlab((s + 1) * 32, buf ^ 1);
      WAIT_ASYNC(9);                 // slab s complete; slab s+1 may be in flight
    } else {
      WAIT_ASYNC(0);
    }
    __syncthreads();                 // slab s visible to all waves

    const float* bufA = &smemA[buf * SLABA];
    const float* bufB = &smemB[buf * SLABB];

    v16h a[2];
#pragma unroll
    for (int rt = 0; rt < 2; ++rt)
#pragma unroll
      for (int e = 0; e < 16; ++e)
        a[rt][e] = (_Float16)bufA[(rt * 16 + l15) * AROW + kbaseA + e + (e & 8)];
    v16h bm[2];
#pragma unroll
    for (int ct = 0; ct < 2; ++ct)
#pragma unroll
      for (int e = 0; e < 16; ++e)
        bm[ct][e] = (_Float16)bufB[(kbaseB + e + (e & 8)) * BROW + lc[ct]];
#pragma unroll
    for (int rt = 0; rt < 2; ++rt)
#pragma unroll
      for (int ct = 0; ct < 2; ++ct)
        acc[rt][ct] = __builtin_amdgcn_wmma_f32_16x16x32_f16(
            false, a[rt], false, bm[ct], (short)0, acc[rt][ct], false, false);

    __syncthreads();                 // all waves done with buf before restage
  }

  // ---- Spill accumulators to LDS (C/D layout: VGPR r -> row r or r+8) -----
#pragma unroll
  for (int rt = 0; rt < 2; ++rt)
#pragma unroll
    for (int ct = 0; ct < 2; ++ct)
#pragma unroll
      for (int r = 0; r < 8; ++r) {
        int row = rt * 16 + r + (hi16 ? 8 : 0);
        gsm[row * (4 * UT) + lc[ct]] = acc[rt][ct][r];
      }
  __syncthreads();

  // ---- LSTM cell update (masked), h double-buffered -----------------------
  for (int i = threadIdx.x; i < BATCHN * UT; i += blockDim.x) {
    int m = i / UT, j = i % UT;
    size_t idx = (size_t)m * HID + u0 + j;
    int tp = tok[m * SEQT + (t - 1)];

    float gi = gsm[m * 256 + j]       + bias[0 * HID + u0 + j];
    float gf = gsm[m * 256 + 64 + j]  + bias[1 * HID + u0 + j];
    float gg = gsm[m * 256 + 128 + j] + bias[2 * HID + u0 + j];
    float go = gsm[m * 256 + 192 + j] + bias[3 * HID + u0 + j];

    if (tp != 0) {
      float ii = 1.0f / (1.0f + expf(-gi));
      float ff = 1.0f / (1.0f + expf(-gf));
      float gt = tanhf(gg);
      float oo = 1.0f / (1.0f + expf(-go));
      float cn = ff * c[idx] + ii * gt;
      float hn = oo * tanhf(cn);
      c[idx]    = cn;
      hout[idx] = hn;
    } else {
      hout[idx] = hin[idx];          // masked: state carried through
    }
  }
}

// ---------------------------------------------------------------------------
// Softmax stats + arithmetic-coding z. One block per batch row.
// z[b,t,:] = 3*cl + 1.5*p_s,  cl = sum_{j<s} softmax(h)[j].
// ---------------------------------------------------------------------------
__global__ void dariel_softmax_z_kernel(const float* __restrict__ h,
                                        const int*   __restrict__ tok,
                                        float*       __restrict__ out,
                                        int t) {
  __shared__ float red[256];
  const int b   = blockIdx.x;
  const int tid = threadIdx.x;
  const float* hr = h + (size_t)b * HID;
  const int s = tok[b * SEQT + t];

  float mx = -3.402823466e38f;
  for (int j = tid; j < HID; j += 256) mx = fmaxf(mx, hr[j]);
  red[tid] = mx; __syncthreads();
  for (int off = 128; off > 0; off >>= 1) {
    if (tid < off) red[tid] = fmaxf(red[tid], red[tid + off]);
    __syncthreads();
  }
  mx = red[0]; __syncthreads();

  float sum = 0.0f, below = 0.0f;
  for (int j = tid; j < HID; j += 256) {
    float e = expf(hr[j] - mx);
    sum += e;
    if (j < s) below += e;
  }
  red[tid] = sum; __syncthreads();
  for (int off = 128; off > 0; off >>= 1) {
    if (tid < off) red[tid] += red[tid + off];
    __syncthreads();
  }
  sum = red[0]; __syncthreads();

  red[tid] = below; __syncthreads();
  for (int off = 128; off > 0; off >>= 1) {
    if (tid < off) red[tid] += red[tid + off];
    __syncthreads();
  }
  below = red[0];

  float ps = expf(hr[s] - mx);
  float z  = 3.0f * (below / sum) + 1.5f * (ps / sum);
  if (tid < LATN)
    out[(size_t)b * SEQT * LATN + (size_t)t * LATN + tid] = z;
}

// ---------------------------------------------------------------------------
extern "C" void kernel_launch(void* const* d_in, const int* in_sizes, int n_in,
                              void* d_out, int out_size, void* d_ws, size_t ws_size,
                              hipStream_t stream) {
  const int*   tok  = (const int*)  d_in[0];   // [32, 8]
  const float* E    = (const float*)d_in[1];   // [2048, 256]
  const float* Wi   = (const float*)d_in[2];   // [256, 8192]
  const float* Wh   = (const float*)d_in[3];   // [2048, 8192]
  const float* bias = (const float*)d_in[4];   // [8192]
  float* out = (float*)d_out;                  // [32, 8, 128]

  float* h0 = (float*)d_ws;                    // 256 KB
  float* h1 = h0 + BATCHN * HID;               // 256 KB
  float* c  = h1 + BATCHN * HID;               // 256 KB
  float* hb[2] = { h0, h1 };

  dariel_init_kernel<<<(BATCHN * HID + 255) / 256, 256, 0, stream>>>(h0, c, out);

  for (int t = 1; t < SEQT; ++t) {
    dariel_lstm_step_kernel<<<HID / UT, 256, 0, stream>>>(
        tok, E, Wi, Wh, bias, hb[(t - 1) & 1], hb[t & 1], c, t);
    dariel_softmax_z_kernel<<<BATCHN, 256, 0, stream>>>(hb[t & 1], tok, out, t);
  }
}
